// DeepSeekV3MoE_18442589569146
// MI455X (gfx1250) — matrix operand order repaired and checked by reading, not compiled, not run on hardware
//
#include <hip/hip_runtime.h>

// ---------------------------------------------------------------------------
// DeepSeekV3 MoE for MI455X (gfx1250, wave32, WMMA bf16 16x16x32, f32 accum)
// GEMM hot loop: async global->LDS DMA (ASYNCcnt) double-buffered against
// ds_load_b128 -> v_wmma_f32_16x16x32_bf16 compute.
// ---------------------------------------------------------------------------

typedef __attribute__((ext_vector_type(16))) __bf16 v16bf;
typedef __attribute__((ext_vector_type(8)))  float  v8f;

#define BM 128
#define BN 128
#define BK 32

union FragBF {
    v16bf v;
    uint4 q[2];
};

__device__ __forceinline__ unsigned short f2bf(float f) {
    unsigned u = __float_as_uint(f);
    u += 0x7FFFu + ((u >> 16) & 1u);   // round-to-nearest-even
    return (unsigned short)(u >> 16);
}

__device__ __forceinline__ float fast_silu(float g) {
    return g * __builtin_amdgcn_rcpf(1.0f + __expf(-g));
}

// low 32 bits of generic pointer == LDS byte offset (LDS aperture mapping)
__device__ __forceinline__ unsigned lds_addr(const void* p) {
    return (unsigned)(unsigned long long)p;
}

// 32 bytes (one tile-row half) global -> LDS, async, per lane
__device__ __forceinline__ void async_copy_32B(unsigned lds, unsigned off,
                                               const unsigned short* base) {
    asm volatile("global_load_async_to_lds_b128 %0, %1, %2\n\t"
                 "global_load_async_to_lds_b128 %0, %1, %2 offset:16"
                 :: "v"(lds), "v"(off), "s"(base) : "memory");
}

__device__ __forceinline__ void wait_async0() {
    asm volatile("s_wait_asynccnt 0x0" ::: "memory");
}

// ---------------------------------------------------------------------------
// f32 -> bf16 elementwise convert (grid-stride), for x
// ---------------------------------------------------------------------------
__global__ void cvt_f32_bf16_kernel(const float* __restrict__ in,
                                    unsigned short* __restrict__ out,
                                    long long n) {
    long long i = (long long)blockIdx.x * blockDim.x + threadIdx.x;
    long long stride = (long long)gridDim.x * blockDim.x;
    for (; i < n; i += stride) out[i] = f2bf(in[i]);
}

// ---------------------------------------------------------------------------
// f32 [K][N] -> bf16 [N][K] tiled transpose+convert (32x32 tiles via LDS)
// ---------------------------------------------------------------------------
__global__ __launch_bounds__(256)
void transpose_cvt_kernel(const float* __restrict__ in,     // [K][N]
                          unsigned short* __restrict__ out, // [N][K]
                          int K, int N) {
    __shared__ unsigned short tile[32][33];
    const int k0 = blockIdx.y * 32;
    const int n0 = blockIdx.x * 32;
    const int tx = threadIdx.x;        // 0..31
    const int ty = threadIdx.y;        // 0..7
#pragma unroll
    for (int r = ty; r < 32; r += 8)
        tile[r][tx] = f2bf(in[(long long)(k0 + r) * N + n0 + tx]);
    __syncthreads();
#pragma unroll
    for (int r = ty; r < 32; r += 8)
        out[(long long)(n0 + r) * K + k0 + tx] = tile[tx][r];
}

// ---------------------------------------------------------------------------
// Sigmoid router + top-2 + normalize -> dense combine[T, 8]; one wave32/token
// ---------------------------------------------------------------------------
__global__ __launch_bounds__(128)
void router_kernel(const float* __restrict__ X,       // [T, D]
                   const float* __restrict__ Wr,      // [D, 8]
                   float* __restrict__ Combine,       // [T, 8]
                   int T, int D) {
    const int lane  = threadIdx.x & 31;
    const int token = blockIdx.x * (blockDim.x >> 5) + (threadIdx.x >> 5);
    if (token >= T) return;

    float acc[8];
#pragma unroll
    for (int e = 0; e < 8; ++e) acc[e] = 0.0f;

    const float* xr = X + (long long)token * D;
    for (int d = lane; d < D; d += 32) {
        float xv = xr[d];
        const float* w = Wr + (long long)d * 8;
#pragma unroll
        for (int e = 0; e < 8; ++e) acc[e] += xv * w[e];
    }
#pragma unroll
    for (int e = 0; e < 8; ++e) {
#pragma unroll
        for (int off = 16; off > 0; off >>= 1)
            acc[e] += __shfl_xor(acc[e], off, 32);
    }
    if (lane == 0) {
        float s[8];
#pragma unroll
        for (int e = 0; e < 8; ++e)
            s[e] = __builtin_amdgcn_rcpf(1.0f + __expf(-acc[e]));
        int i0 = 0;
#pragma unroll
        for (int e = 1; e < 8; ++e) if (s[e] > s[i0]) i0 = e;
        int i1 = (i0 == 0) ? 1 : 0;
#pragma unroll
        for (int e = 0; e < 8; ++e) if (e != i0 && s[e] > s[i1]) i1 = e;
        float inv = __builtin_amdgcn_rcpf(s[i0] + s[i1]);
        float* c = Combine + (long long)token * 8;
#pragma unroll
        for (int e = 0; e < 8; ++e)
            c[e] = (e == i0) ? s[i0] * inv : ((e == i1) ? s[i1] * inv : 0.0f);
    }
}

// ---------------------------------------------------------------------------
// Fused gate+up: H = silu(A @ G^T) * (A @ U^T), all bf16 operands.
// A bf16 [M][K]; Gt/Ut bf16 [N][K] (k-contiguous).
// Block 128x128, BK=32, 256 threads = 8 waves (4x2), wave tile 32x64.
// Double-buffered async global->LDS pipeline.
// ---------------------------------------------------------------------------
__global__ __launch_bounds__(256)
void moe_gateup_kernel(const unsigned short* __restrict__ A,   // [M][K]
                       const unsigned short* __restrict__ Gt,  // [N][K]
                       const unsigned short* __restrict__ Ut,  // [N][K]
                       unsigned short* __restrict__ H,         // [M][N]
                       int Kdim, int N) {
    __shared__ __align__(16) unsigned short As[2][BM][BK];
    __shared__ __align__(16) unsigned short Bg[2][BN][BK];
    __shared__ __align__(16) unsigned short Bu[2][BN][BK];

    const int tid  = threadIdx.x;
    const int lane = tid & 31;
    const int wave = tid >> 5;
    const int wm   = wave & 3;        // 0..3
    const int wn   = wave >> 2;       // 0..1
    const int m0   = blockIdx.y * BM;
    const int n0   = blockIdx.x * BN;
    const int lg   = lane >> 4;
    const int ml   = lane & 15;

    v8f accG[2][4] = {};
    v8f accU[2][4] = {};

    const int srow = tid >> 1;          // 0..127
    const int scol = (tid & 1) * 16;    // 0 / 16

    unsigned offA = (unsigned)((m0 + srow) * Kdim + scol) * 2u;
    unsigned offB = (unsigned)((n0 + srow) * Kdim + scol) * 2u;
    unsigned ldsA[2], ldsG[2], ldsU[2];
#pragma unroll
    for (int b = 0; b < 2; ++b) {
        ldsA[b] = lds_addr(&As[b][srow][scol]);
        ldsG[b] = lds_addr(&Bg[b][srow][scol]);
        ldsU[b] = lds_addr(&Bu[b][srow][scol]);
    }

    // prologue: stage tile 0 into buffer 0
    async_copy_32B(ldsA[0], offA, A);
    async_copy_32B(ldsG[0], offB, Gt);
    async_copy_32B(ldsU[0], offB, Ut);
    offA += BK * 2;  offB += BK * 2;

    const int nK = Kdim / BK;
    for (int kk = 0; kk < nK; ++kk) {
        const int buf = kk & 1;
        wait_async0();        // this wave's tile-kk writes complete
        __syncthreads();      // all waves' tile-kk writes complete; prev reads done
        if (kk + 1 < nK) {    // prefetch tile kk+1 into the other buffer
            const int nb = buf ^ 1;
            async_copy_32B(ldsA[nb], offA, A);
            async_copy_32B(ldsG[nb], offB, Gt);
            async_copy_32B(ldsU[nb], offB, Ut);
            offA += BK * 2;  offB += BK * 2;
        }

        FragBF a[2];
#pragma unroll
        for (int i = 0; i < 2; ++i) {
            const int row = wm * 32 + i * 16 + ml;
            a[i].q[0] = *reinterpret_cast<const uint4*>(&As[buf][row][lg * 16]);
            a[i].q[1] = *reinterpret_cast<const uint4*>(&As[buf][row][lg * 16 + 8]);
        }
#pragma unroll
        for (int j = 0; j < 4; ++j) {
            const int nrow = wn * 64 + j * 16 + ml;
            FragBF bg, bu;
            bg.q[0] = *reinterpret_cast<const uint4*>(&Bg[buf][nrow][lg * 16]);
            bg.q[1] = *reinterpret_cast<const uint4*>(&Bg[buf][nrow][lg * 16 + 8]);
            bu.q[0] = *reinterpret_cast<const uint4*>(&Bu[buf][nrow][lg * 16]);
            bu.q[1] = *reinterpret_cast<const uint4*>(&Bu[buf][nrow][lg * 16 + 8]);
#pragma unroll
            for (int i = 0; i < 2; ++i) {
                accG[i][j] = __builtin_amdgcn_wmma_f32_16x16x32_bf16(
                    false, a[i].v, false, bg.v, (short)0, accG[i][j], false, false);
                accU[i][j] = __builtin_amdgcn_wmma_f32_16x16x32_bf16(
                    false, a[i].v, false, bu.v, (short)0, accU[i][j], false, false);
            }
        }
    }

    // SwiGLU epilogue -> bf16 H (incremental addressing, no per-elem mul64)
#pragma unroll
    for (int i = 0; i < 2; ++i) {
        const int mb = m0 + wm * 32 + i * 16 + lg * 8;
#pragma unroll
        for (int j = 0; j < 4; ++j) {
            const int nn = n0 + wn * 64 + j * 16 + ml;
            unsigned short* hp = H + (long long)mb * N + nn;
#pragma unroll
            for (int r = 0; r < 8; ++r) {
                *hp = f2bf(fast_silu(accG[i][j][r]) * accU[i][j][r]);
                hp += N;
            }
        }
    }
}

// ---------------------------------------------------------------------------
// Down projection with per-token routing weight:
//   Out[m,n] += scale[m*stride] * (A @ Bt^T)[m,n]   (scale==nullptr -> 1.0)
// ---------------------------------------------------------------------------
__global__ __launch_bounds__(256)
void moe_down_kernel(const unsigned short* __restrict__ A,    // [M][K]
                     const unsigned short* __restrict__ Bt,   // [N][K]
                     const float* __restrict__ scale,
                     int scaleStride,
                     float* __restrict__ Out,                 // [M][N]
                     int Kdim, int N) {
    __shared__ __align__(16) unsigned short As[2][BM][BK];
    __shared__ __align__(16) unsigned short Bs[2][BN][BK];

    const int tid  = threadIdx.x;
    const int lane = tid & 31;
    const int wave = tid >> 5;
    const int wm   = wave & 3;
    const int wn   = wave >> 2;
    const int m0   = blockIdx.y * BM;
    const int n0   = blockIdx.x * BN;
    const int lg   = lane >> 4;
    const int ml   = lane & 15;

    v8f acc[2][4] = {};

    const int srow = tid >> 1;
    const int scol = (tid & 1) * 16;

    unsigned offA = (unsigned)((m0 + srow) * Kdim + scol) * 2u;
    unsigned offB = (unsigned)((n0 + srow) * Kdim + scol) * 2u;
    unsigned ldsA[2], ldsB[2];
#pragma unroll
    for (int b = 0; b < 2; ++b) {
        ldsA[b] = lds_addr(&As[b][srow][scol]);
        ldsB[b] = lds_addr(&Bs[b][srow][scol]);
    }

    async_copy_32B(ldsA[0], offA, A);
    async_copy_32B(ldsB[0], offB, Bt);
    offA += BK * 2;  offB += BK * 2;

    const int nK = Kdim / BK;
    for (int kk = 0; kk < nK; ++kk) {
        const int buf = kk & 1;
        wait_async0();
        __syncthreads();
        if (kk + 1 < nK) {
            const int nb = buf ^ 1;
            async_copy_32B(ldsA[nb], offA, A);
            async_copy_32B(ldsB[nb], offB, Bt);
            offA += BK * 2;  offB += BK * 2;
        }

        FragBF a[2];
#pragma unroll
        for (int i = 0; i < 2; ++i) {
            const int row = wm * 32 + i * 16 + ml;
            a[i].q[0] = *reinterpret_cast<const uint4*>(&As[buf][row][lg * 16]);
            a[i].q[1] = *reinterpret_cast<const uint4*>(&As[buf][row][lg * 16 + 8]);
        }
#pragma unroll
        for (int j = 0; j < 4; ++j) {
            const int nrow = wn * 64 + j * 16 + ml;
            FragBF b;
            b.q[0] = *reinterpret_cast<const uint4*>(&Bs[buf][nrow][lg * 16]);
            b.q[1] = *reinterpret_cast<const uint4*>(&Bs[buf][nrow][lg * 16 + 8]);
#pragma unroll
            for (int i = 0; i < 2; ++i)
                acc[i][j] = __builtin_amdgcn_wmma_f32_16x16x32_bf16(
                    false, a[i].v, false, b.v, (short)0, acc[i][j], false, false);
        }
    }

    // scaled accumulate epilogue (incremental addressing)
#pragma unroll
    for (int i = 0; i < 2; ++i) {
        const int mb = m0 + wm * 32 + i * 16 + lg * 8;
        float sc[8];
        if (scale) {
            const float* sp = scale + (long long)mb * scaleStride;
#pragma unroll
            for (int r = 0; r < 8; ++r) { sc[r] = *sp; sp += scaleStride; }
        } else {
#pragma unroll
            for (int r = 0; r < 8; ++r) sc[r] = 1.0f;
        }
#pragma unroll
        for (int j = 0; j < 4; ++j) {
            const int nn = n0 + wn * 64 + j * 16 + ml;
            float* op = Out + (long long)mb * N + nn;
#pragma unroll
            for (int r = 0; r < 8; ++r) {
                *op += sc[r] * acc[i][j][r];
                op += N;
            }
        }
    }
}

// ---------------------------------------------------------------------------
// Host launch
// ---------------------------------------------------------------------------
extern "C" void kernel_launch(void* const* d_in, const int* in_sizes, int n_in,
                              void* d_out, int out_size, void* d_ws, size_t ws_size,
                              hipStream_t stream) {
    constexpr int Dd = 2048, Ec = 8, Fc = 1408, FSc = 2816;
    constexpr int T  = 4 * 2048;   // B*S = 8192

    const float* x   = (const float*)d_in[0];
    const float* wr  = (const float*)d_in[1];
    const float* w1  = (const float*)d_in[2];   // [E, D, F]
    const float* w2  = (const float*)d_in[3];   // [E, F, D]
    const float* w3  = (const float*)d_in[4];   // [E, D, F]
    const float* ws1 = (const float*)d_in[5];   // [D, FS]
    const float* ws2 = (const float*)d_in[6];   // [FS, D]
    const float* ws3 = (const float*)d_in[7];   // [D, FS]
    float* out = (float*)d_out;

    char* ws = (char*)d_ws;
    size_t off = 0;
    auto carve = [&](size_t bytes) { void* p = ws + off; off += bytes; return p; };

    unsigned short* xb      = (unsigned short*)carve((size_t)T * Dd * 2);
    float*          combine = (float*)carve((size_t)T * Ec * 4);
    unsigned short* h       = (unsigned short*)carve((size_t)T * FSc * 2);
    unsigned short* w1t     = (unsigned short*)carve((size_t)Ec * Fc * Dd * 2);
    unsigned short* w3t     = (unsigned short*)carve((size_t)Ec * Fc * Dd * 2);
    unsigned short* w2t     = (unsigned short*)carve((size_t)Ec * Dd * Fc * 2);
    unsigned short* ws1t    = (unsigned short*)carve((size_t)FSc * Dd * 2);
    unsigned short* ws3t    = (unsigned short*)carve((size_t)FSc * Dd * 2);
    unsigned short* ws2t    = (unsigned short*)carve((size_t)Dd * FSc * 2);

    const dim3 tb(32, 8);

    cvt_f32_bf16_kernel<<<4096, 256, 0, stream>>>(x, xb, (long long)T * Dd);

    for (int e = 0; e < Ec; ++e) {
        transpose_cvt_kernel<<<dim3(Fc / 32, Dd / 32), tb, 0, stream>>>(
            w1 + (size_t)e * Dd * Fc, w1t + (size_t)e * Fc * Dd, Dd, Fc);
        transpose_cvt_kernel<<<dim3(Fc / 32, Dd / 32), tb, 0, stream>>>(
            w3 + (size_t)e * Dd * Fc, w3t + (size_t)e * Fc * Dd, Dd, Fc);
        transpose_cvt_kernel<<<dim3(Dd / 32, Fc / 32), tb, 0, stream>>>(
            w2 + (size_t)e * Fc * Dd, w2t + (size_t)e * Dd * Fc, Fc, Dd);
    }
    transpose_cvt_kernel<<<dim3(FSc / 32, Dd / 32), tb, 0, stream>>>(ws1, ws1t, Dd, FSc);
    transpose_cvt_kernel<<<dim3(FSc / 32, Dd / 32), tb, 0, stream>>>(ws3, ws3t, Dd, FSc);
    transpose_cvt_kernel<<<dim3(Dd / 32, FSc / 32), tb, 0, stream>>>(ws2, ws2t, FSc, Dd);

    router_kernel<<<T / 4, 128, 0, stream>>>(x, wr, combine, T, Dd);

    hipMemsetAsync(d_out, 0, (size_t)T * Dd * sizeof(float), stream);

    const dim3 blk(256);
    for (int e = 0; e < Ec; ++e) {
        moe_gateup_kernel<<<dim3(Fc / BN, T / BM), blk, 0, stream>>>(
            xb, w1t + (size_t)e * Fc * Dd, w3t + (size_t)e * Fc * Dd, h, Dd, Fc);
        moe_down_kernel<<<dim3(Dd / BN, T / BM), blk, 0, stream>>>(
            h, w2t + (size_t)e * Dd * Fc, combine + e, Ec, out, Fc, Dd);
    }

    moe_gateup_kernel<<<dim3(FSc / BN, T / BM), blk, 0, stream>>>(
        xb, ws1t, ws3t, h, Dd, FSc);
    moe_down_kernel<<<dim3(Dd / BN, T / BM), blk, 0, stream>>>(
        h, ws2t, nullptr, 0, out, FSc, Dd);
}
